// HyperNetLinear_42520176230701
// MI455X (gfx1250) — compile-verified, matrix operand order checked by reading, not run on hardware
//
#include <hip/hip_runtime.h>

#define BN   8192
#define INN  256
#define OUTN 256
#define LATN 128
#define KTOT (LATN * INN)     // 32768 : K of the fused bilinear GEMM

typedef __bf16 bf16;
typedef __attribute__((ext_vector_type(2)))  bf16     bf16x2;
typedef __attribute__((ext_vector_type(16))) bf16     v16bf;
typedef __attribute__((ext_vector_type(8)))  float    v8f;
typedef __attribute__((ext_vector_type(4)))  float    f32x4;
typedef __attribute__((ext_vector_type(4)))  unsigned u32x4;

union Frag16 {
  v16bf    v;
  u32x4    q[2];
  unsigned u[8];
};

// ---- tiling ----
#define M_TILE   128          // rows per block  -> 64 blocks
#define K_STEP   32           // K slab: latent index l is constant per slab
#define THREADS  512          // 16 wave32s
#define LAT_PAD  136          // bf16 per latent row (bank-conflict pad)
#define XCH_PAD  40           // bf16 per x-chunk row (16B-aligned frags)
#define BPAD     40           // bf16 per B column (K-major, 16B-aligned)
#define LAT_ELE  (M_TILE * LAT_PAD)
#define XCH_ELE  (M_TILE * XCH_PAD)
#define BBUF_ELE (OUTN * BPAD)
#define SMEM_BYTES ((LAT_ELE + XCH_ELE + 2 * BBUF_ELE) * 2)   // 86016 B

// ---------------------------------------------------------------------------
// out[b,o] = x[b,:]@bw[:,o] + latent[b,:]@Wb[:,o] + bb[o]   (initializes out)
// ---------------------------------------------------------------------------
__global__ __launch_bounds__(OUTN) void hyper_bias(
    const float* __restrict__ x, const float* __restrict__ latent,
    const float* __restrict__ bw, const float* __restrict__ Wb,
    const float* __restrict__ bb, float* __restrict__ out)
{
  __shared__ float xs[INN];
  __shared__ float ls[LATN];
  const int b = blockIdx.x;
  const int t = threadIdx.x;
  xs[t] = x[(size_t)b * INN + t];
  if (t < LATN) ls[t] = latent[(size_t)b * LATN + t];
  __syncthreads();
  float acc = bb[t];
#pragma unroll 4
  for (int i = 0; i < INN; ++i)  acc = fmaf(xs[i], bw[(size_t)i * OUTN + t], acc);
#pragma unroll 4
  for (int l = 0; l < LATN; ++l) acc = fmaf(ls[l], Wb[(size_t)l * OUTN + t], acc);
  out[(size_t)b * OUTN + t] = acc;
}

// ---------------------------------------------------------------------------
// Prep: Ww fp32 [K=32768][N=256]  ->  WwT bf16 [N=256][K=32768] (in d_ws).
// 64-wide K chunks, transpose through LDS; reads coalesced, writes 128B runs.
// ---------------------------------------------------------------------------
#define TK 64
__global__ __launch_bounds__(256) void ww_to_bf16_t(
    const float* __restrict__ Ww, bf16* __restrict__ WwT)
{
  __shared__ bf16 tile[OUTN * (TK + 8)];        // [n][k], pad 8 (row=144B)
  const int t  = threadIdx.x;
  const int k0 = blockIdx.x * TK;
  for (int idx = t; idx < TK * OUTN; idx += 256) {
    int k = idx >> 8, n = idx & 255;
    tile[n * (TK + 8) + k] = (bf16)Ww[(size_t)(k0 + k) * OUTN + n];
  }
  __syncthreads();
  const u32x4* src = (const u32x4*)(tile + t * (TK + 8));
  u32x4*       dst = (u32x4*)(WwT + (size_t)t * KTOT + k0);
#pragma unroll
  for (int i = 0; i < (TK * 2 / 16); ++i) dst[i] = src[i];
}

// ---------------------------------------------------------------------------
// Fast path: bilinear GEMM streaming pre-converted bf16 WwT slabs straight
// into LDS with GLOBAL_LOAD_ASYNC_TO_LDS_B128 (ASYNCcnt double-buffering).
// ---------------------------------------------------------------------------
__global__ __launch_bounds__(THREADS) void hyper_gemm_async(
    const float* __restrict__ x, const float* __restrict__ latent,
    const bf16* __restrict__ WwT, float* __restrict__ out)
{
  extern __shared__ char smem_raw[];
  bf16* lat_s = (bf16*)smem_raw;            // [128][LAT_PAD]
  bf16* xch_s = lat_s + LAT_ELE;            // [128][XCH_PAD]
  bf16* bbuf  = xch_s + XCH_ELE;            // 2 x [256][BPAD] (K-major)

  const int t    = threadIdx.x;
  const int lane = t & 31;
  const int w    = t >> 5;
  const int wr   = w >> 2;        // 0..3 : 32-row group
  const int wc   = w & 3;         // 0..3 : 64-col group
  const int mrow = lane & 15;
  const int half = lane >> 4;
  const int hs8  = half << 3;
  const int row0 = blockIdx.x * M_TILE;

  for (int idx = t; idx < M_TILE * LATN; idx += THREADS) {
    int r = idx >> 7, c = idx & (LATN - 1);
    lat_s[r * LAT_PAD + c] = (bf16)latent[(size_t)(row0 + r) * LATN + c];
  }

  v8f acc[2][4] = {};

  // async-copy assignment: thread t owns column n, k-half ak (16 bf16 = 32B)
  const int an = t & 255;
  const int ak = (t >> 8) << 4;                         // 0 | 16
  const unsigned lds_b0 =
      (unsigned)(unsigned long long)(bbuf + an * BPAD + ak);
  const unsigned lds_b1 =
      (unsigned)(unsigned long long)(bbuf + BBUF_ELE + an * BPAD + ak);
  const int gvo = an * (KTOT * 2) + ak * 2;             // per-lane byte offset

  for (int c8 = 0; c8 < 8; ++c8) {
    // kick off async copy of slab (l=0, this i-chunk) into buffer 0
    {
      unsigned long long sa =
          (unsigned long long)(WwT + (size_t)(c8 << 5));
      asm volatile("global_load_async_to_lds_b128 %0, %1, %2"
                   :: "v"(lds_b0), "v"(gvo), "s"(sa) : "memory");
      asm volatile("global_load_async_to_lds_b128 %0, %1, %2 offset:16"
                   :: "v"(lds_b0), "v"(gvo), "s"(sa) : "memory");
    }
    // stage x chunk [128][32] fp32 -> bf16 (overlaps with async copy)
    for (int idx = t; idx < M_TILE * K_STEP; idx += THREADS) {
      int r = idx >> 5, c = idx & 31;
      xch_s[r * XCH_PAD + c] = (bf16)x[(size_t)(row0 + r) * INN + (c8 << 5) + c];
    }
    asm volatile("s_wait_asynccnt 0x0" ::: "memory");
    __syncthreads();

    // hoist x fragments (constant across the 128 l-iterations)
    u32x4 ax[2][2];
#pragma unroll
    for (int rt = 0; rt < 2; ++rt) {
      int r = wr * 32 + rt * 16 + mrow;
      ax[rt][0] = *(const u32x4*)(xch_s + r * XCH_PAD + hs8);
      ax[rt][1] = *(const u32x4*)(xch_s + r * XCH_PAD + 16 + hs8);
    }

    int cur = 0;
    for (int l = 0; l < LATN; ++l) {
      // async copy of next slab into the other buffer
      if (l + 1 < LATN) {
        unsigned long long sa = (unsigned long long)
            (WwT + ((size_t)(l + 1) * INN + (c8 << 5)));
        unsigned ld = cur ? lds_b0 : lds_b1;
        asm volatile("global_load_async_to_lds_b128 %0, %1, %2"
                     :: "v"(ld), "v"(gvo), "s"(sa) : "memory");
        asm volatile("global_load_async_to_lds_b128 %0, %1, %2 offset:16"
                     :: "v"(ld), "v"(gvo), "s"(sa) : "memory");
        if (l + 2 < LATN)   // warm L2 one more slab ahead
          __builtin_prefetch(
              (const char*)(WwT + ((size_t)(l + 2) * INN + (c8 << 5))) + gvo,
              0, 1);
      }
      // A fragments: hoisted x slice * latent scalar (packed bf16 mul)
      Frag16 af[2];
#pragma unroll
      for (int rt = 0; rt < 2; ++rt) {
        int r = wr * 32 + rt * 16 + mrow;
        unsigned short lb =
            __builtin_bit_cast(unsigned short, lat_s[r * LAT_PAD + l]);
        unsigned lpu = (unsigned)lb * 0x00010001u;
        af[rt].q[0] = ax[rt][0];
        af[rt].q[1] = ax[rt][1];
#pragma unroll
        for (int j = 0; j < 8; ++j) {
          unsigned prod;
          asm("v_pk_mul_bf16 %0, %1, %2"
              : "=v"(prod) : "v"(af[rt].u[j]), "v"(lpu));
          af[rt].u[j] = prod;
        }
      }
      const bf16* bsr = cur ? (bbuf + BBUF_ELE) : bbuf;
#pragma unroll
      for (int ct = 0; ct < 4; ++ct) {
        int n = wc * 64 + ct * 16 + mrow;
        Frag16 bfr;
        bfr.q[0] = *(const u32x4*)(bsr + n * BPAD + hs8);
        bfr.q[1] = *(const u32x4*)(bsr + n * BPAD + 16 + hs8);
#pragma unroll
        for (int rt = 0; rt < 2; ++rt) {
          acc[rt][ct] = __builtin_amdgcn_wmma_f32_16x16x32_bf16(
              false, af[rt].v, false, bfr.v, (short)0, acc[rt][ct],
              false, false);
        }
      }
      asm volatile("s_wait_asynccnt 0x0" ::: "memory");
      __syncthreads();
      cur ^= 1;
    }
    __syncthreads();
  }

#pragma unroll
  for (int rt = 0; rt < 2; ++rt) {
#pragma unroll
    for (int ct = 0; ct < 4; ++ct) {
      int col   = wc * 64 + ct * 16 + mrow;
      int rbase = row0 + wr * 32 + rt * 16 + (half ? 8 : 0);
#pragma unroll
      for (int j = 0; j < 8; ++j) {
        size_t o = (size_t)(rbase + j) * OUTN + col;
        out[o] += acc[rt][ct][j];
      }
    }
  }
}

// ---------------------------------------------------------------------------
// Fallback (no workspace): fp32 Ww loads + on-the-fly bf16 convert via LDS.
// ---------------------------------------------------------------------------
__global__ __launch_bounds__(THREADS) void hyper_gemm(
    const float* __restrict__ x, const float* __restrict__ latent,
    const float* __restrict__ Ww, float* __restrict__ out)
{
  extern __shared__ char smem_raw[];
  bf16* lat_s = (bf16*)smem_raw;
  bf16* xch_s = lat_s + LAT_ELE;
  bf16* bbuf  = xch_s + XCH_ELE;

  const int t    = threadIdx.x;
  const int lane = t & 31;
  const int w    = t >> 5;
  const int wr   = w >> 2;
  const int wc   = w & 3;
  const int mrow = lane & 15;
  const int half = lane >> 4;
  const int hs8  = half << 3;
  const int row0 = blockIdx.x * M_TILE;

  for (int idx = t; idx < M_TILE * LATN; idx += THREADS) {
    int r = idx >> 7, c = idx & (LATN - 1);
    lat_s[r * LAT_PAD + c] = (bf16)latent[(size_t)(row0 + r) * LATN + c];
  }

  v8f acc[2][4] = {};
  f32x4 ga[2], gb[2];
  const int n0 = (t & 63) << 2;
  const int kp = ((t >> 6) & 7) << 1;

  for (int c8 = 0; c8 < 8; ++c8) {
    for (int idx = t; idx < M_TILE * K_STEP; idx += THREADS) {
      int r = idx >> 5, c = idx & 31;
      xch_s[r * XCH_PAD + c] = (bf16)x[(size_t)(row0 + r) * INN + (c8 << 5) + c];
    }
    {
      const float* src = Ww + (size_t)(c8 << 5) * OUTN;
#pragma unroll
      for (int p = 0; p < 2; ++p) {
        int k = kp + (p << 4);
        ga[p] = *(const f32x4*)(src + k * OUTN + n0);
        gb[p] = *(const f32x4*)(src + (k + 1) * OUTN + n0);
      }
    }
    __syncthreads();

    u32x4 ax[2][2];
#pragma unroll
    for (int rt = 0; rt < 2; ++rt) {
      int r = wr * 32 + rt * 16 + mrow;
      ax[rt][0] = *(const u32x4*)(xch_s + r * XCH_PAD + hs8);
      ax[rt][1] = *(const u32x4*)(xch_s + r * XCH_PAD + 16 + hs8);
    }

    int cur = 0;
    for (int l = 0; l < LATN; ++l) {
      bf16* bs = bbuf + cur * BBUF_ELE;
#pragma unroll
      for (int p = 0; p < 2; ++p) {
        int k = kp + (p << 4);
#pragma unroll
        for (int j = 0; j < 4; ++j) {
          bf16x2 pk;
          pk.x = (bf16)ga[p][j];
          pk.y = (bf16)gb[p][j];
          *(bf16x2*)(bs + (n0 + j) * BPAD + k) = pk;
        }
      }
      if (l + 1 < LATN) {
        const float* src = Ww + ((size_t)(l + 1) * INN + (c8 << 5)) * OUTN;
#pragma unroll
        for (int p = 0; p < 2; ++p) {
          int k = kp + (p << 4);
          ga[p] = *(const f32x4*)(src + k * OUTN + n0);
          gb[p] = *(const f32x4*)(src + (k + 1) * OUTN + n0);
        }
        if (l + 2 < LATN)
          __builtin_prefetch(src + (size_t)INN * OUTN + t * 16, 0, 1);
      }
      __syncthreads();

      Frag16 af[2];
#pragma unroll
      for (int rt = 0; rt < 2; ++rt) {
        int r = wr * 32 + rt * 16 + mrow;
        unsigned short lb =
            __builtin_bit_cast(unsigned short, lat_s[r * LAT_PAD + l]);
        unsigned lpu = (unsigned)lb * 0x00010001u;
        af[rt].q[0] = ax[rt][0];
        af[rt].q[1] = ax[rt][1];
#pragma unroll
        for (int j = 0; j < 8; ++j) {
          unsigned prod;
          asm("v_pk_mul_bf16 %0, %1, %2"
              : "=v"(prod) : "v"(af[rt].u[j]), "v"(lpu));
          af[rt].u[j] = prod;
        }
      }
      const bf16* bsr = bbuf + cur * BBUF_ELE;
#pragma unroll
      for (int ct = 0; ct < 4; ++ct) {
        int n = wc * 64 + ct * 16 + mrow;
        Frag16 bfr;
        bfr.q[0] = *(const u32x4*)(bsr + n * BPAD + hs8);
        bfr.q[1] = *(const u32x4*)(bsr + n * BPAD + 16 + hs8);
#pragma unroll
        for (int rt = 0; rt < 2; ++rt) {
          acc[rt][ct] = __builtin_amdgcn_wmma_f32_16x16x32_bf16(
              false, af[rt].v, false, bfr.v, (short)0, acc[rt][ct],
              false, false);
        }
      }
      cur ^= 1;
    }
    __syncthreads();
  }

#pragma unroll
  for (int rt = 0; rt < 2; ++rt) {
#pragma unroll
    for (int ct = 0; ct < 4; ++ct) {
      int col   = wc * 64 + ct * 16 + mrow;
      int rbase = row0 + wr * 32 + rt * 16 + (half ? 8 : 0);
#pragma unroll
      for (int j = 0; j < 8; ++j) {
        size_t o = (size_t)(rbase + j) * OUTN + col;
        out[o] += acc[rt][ct][j];
      }
    }
  }
}

// ---------------------------------------------------------------------------
extern "C" void kernel_launch(void* const* d_in, const int* in_sizes, int n_in,
                              void* d_out, int out_size, void* d_ws,
                              size_t ws_size, hipStream_t stream) {
  (void)in_sizes; (void)n_in; (void)out_size;
  const float* x      = (const float*)d_in[0];
  const float* latent = (const float*)d_in[1];
  const float* Ww     = (const float*)d_in[2];
  const float* bw     = (const float*)d_in[3];
  const float* Wb     = (const float*)d_in[4];
  const float* bb     = (const float*)d_in[5];
  float* out = (float*)d_out;

  hyper_bias<<<dim3(BN), dim3(OUTN), 0, stream>>>(x, latent, bw, Wb, bb, out);

  const size_t WWT_BYTES = (size_t)OUTN * KTOT * sizeof(bf16);   // 16 MB
  if (ws_size >= WWT_BYTES) {
    bf16* WwT = (bf16*)d_ws;
    ww_to_bf16_t<<<dim3(KTOT / TK), dim3(256), 0, stream>>>(Ww, WwT);
    hyper_gemm_async<<<dim3(BN / M_TILE), dim3(THREADS), SMEM_BYTES, stream>>>(
        x, latent, WwT, out);
  } else {
    hyper_gemm<<<dim3(BN / M_TILE), dim3(THREADS), SMEM_BYTES, stream>>>(
        x, latent, Ww, out);
  }
}
